// _GraphTokenEncoder_61203283968404
// MI455X (gfx1250) — matrix operand order repaired and checked by reading, not compiled, not run on hardware
//
#include <hip/hip_runtime.h>
#include <hip/hip_bf16.h>
#include <math.h>

#define N_NODES 100000
#define N_EDGES 640000
#define DIM     128
#define NLAYERS 3

typedef __attribute__((ext_vector_type(16))) __bf16 v16bf;
typedef __attribute__((ext_vector_type(8)))  __bf16 v8bf;
typedef __attribute__((ext_vector_type(4)))  __bf16 v4bf;
typedef __attribute__((ext_vector_type(8)))  float  v8f;

#define WMMA_BF16(a, b, c) \
  __builtin_amdgcn_wmma_f32_16x16x32_bf16(false, (a), false, (b), (short)0, (c), false, false)

// ---------------- WMMA fragment loaders (ISA 7.12.2 layouts) -----------------
// A (16x32 bf16, row-major in LDS, stride ld): lane M = lane&15, half = lane>>4
//   elems 0..7  -> K = 8*half + 0..7        (one ds_load_b128)
//   elems 8..15 -> K = 16 + 8*half + 0..7   (one ds_load_b128)
__device__ __forceinline__ v16bf frag_a(const __bf16* tile, int ld) {
  const int lane = threadIdx.x & 31;
  const int half = lane >> 4, m = lane & 15;
  const __bf16* p = tile + m * ld + 8 * half;
  v8bf lo = *(const v8bf*)(p);
  v8bf hi = *(const v8bf*)(p + 16);
  v16bf a;
#pragma unroll
  for (int i = 0; i < 8; ++i) { a[i] = lo[i]; a[i + 8] = hi[i]; }
  return a;
}

// B (32x16 bf16) from W^T stored [n][k] (K-contiguous, stride ld):
//   lane N = lane&15, half = lane>>4; elems 0..7 -> K = 8*half + e, elems 8..15 -> K = 16 + 8*half + (e-8) (two b128)
__device__ __forceinline__ v16bf frag_b(const __bf16* wt, int ld, int k0, int n0) {
  const int lane = threadIdx.x & 31;
  const int half = lane >> 4, n = lane & 15;
  const __bf16* p = wt + (n0 + n) * ld + k0 + 8 * half;
  v8bf lo = *(const v8bf*)(p);
  v8bf hi = *(const v8bf*)(p + 16);
  v16bf b;
#pragma unroll
  for (int i = 0; i < 8; ++i) { b[i] = lo[i]; b[i + 8] = hi[i]; }
  return b;
}

// C/D frag: elem r -> row (r + 8*half), col (lane&15). Bias varies by column.
__device__ __forceinline__ v8f frag_bias(const float* bias, int n0) {
  const float bv = bias[n0 + (threadIdx.x & 15)];
  v8f c;
#pragma unroll
  for (int r = 0; r < 8; ++r) c[r] = bv;
  return c;
}

__device__ __forceinline__ float gelu_exact(float x) {
  return 0.5f * x * (1.0f + erff(x * 0.7071067811865475f));
}

// ---- one-time: W[k][n] (f32) -> padded transposed bf16 image Wt[n][k], ld ---
__global__ void k_prep_wt(const float* __restrict__ w, __bf16* __restrict__ o,
                          int K, int logN, int ld) {
  int i = blockIdx.x * blockDim.x + threadIdx.x;
  if (i < (K << logN)) {
    int k = i >> logN, n = i & ((1 << logN) - 1);
    o[n * ld + k] = (__bf16)w[i];
  }
}

// ---------------- node = x @ node_w + node_b  ([N,128]x[128,128]) ------------
__global__ __launch_bounds__(128) void k_node_embed(
    const float* __restrict__ x, const __bf16* __restrict__ wt,
    const float* __restrict__ b, float* __restrict__ out, int nrows) {
  __shared__ __bf16 sWt[128 * 136];  // W^T [n][k]
  __shared__ __bf16 sA[64 * 136];
  const int tid = threadIdx.x;
  const int row0 = blockIdx.x * 64;
  for (int i = tid; i < (128 * 136) / 8; i += 128)
    ((v8bf*)sWt)[i] = ((const v8bf*)wt)[i];
  for (int i = tid; i < 64 * 32; i += 128) {
    int r = i >> 5, c4 = (i & 31) << 2;
    int gr = row0 + r; if (gr >= nrows) gr = nrows - 1;
    float4 v = *(const float4*)(x + (size_t)gr * 128 + c4);
    __bf16* d = sA + r * 136 + c4;
    d[0] = (__bf16)v.x; d[1] = (__bf16)v.y; d[2] = (__bf16)v.z; d[3] = (__bf16)v.w;
  }
  __syncthreads();
  const int wave = tid >> 5, lane = tid & 31, half = lane >> 4, n = lane & 15;
  const bool full = (row0 + 64 <= nrows);
  const int rbase = row0 + wave * 16 + half * 8;
  float* outb = out + (size_t)rbase * 128 + n;
  const __bf16* at = sA + wave * 16 * 136;
  v16bf afr[4];
#pragma unroll
  for (int kk = 0; kk < 4; ++kk) afr[kk] = frag_a(at + kk * 32, 136);
#pragma unroll
  for (int nt = 0; nt < 8; ++nt) {
    const int n0 = nt * 16;
    v8f acc = frag_bias(b, n0);
#pragma unroll
    for (int kk = 0; kk < 4; ++kk)
      acc = WMMA_BF16(afr[kk], frag_b(sWt, 136, kk * 32, n0), acc);
    float* po = outb + n0;
    if (full) {
#pragma unroll
      for (int r = 0; r < 8; ++r) po[(size_t)r * 128] = acc[r];
    } else {
#pragma unroll
      for (int r = 0; r < 8; ++r)
        if (rbase + r < nrows) po[(size_t)r * 128] = acc[r];
    }
  }
}

// ------------- edge_emb(bf16) = edge_attr @ edge_w + edge_b  (K=32) ----------
__global__ __launch_bounds__(128) void k_edge_embed(
    const float* __restrict__ ea, const __bf16* __restrict__ wt,
    const float* __restrict__ b, __bf16* __restrict__ out, int nrows) {
  __shared__ __bf16 sWt[128 * 40];  // W^T [n<128][k<32], ld=40
  __shared__ __bf16 sA[64 * 40];
  const int tid = threadIdx.x;
  const int row0 = blockIdx.x * 64;
  for (int i = tid; i < (128 * 40) / 8; i += 128)
    ((v8bf*)sWt)[i] = ((const v8bf*)wt)[i];
  for (int i = tid; i < 64 * 8; i += 128) {
    int r = i >> 3, c4 = (i & 7) << 2;
    int gr = row0 + r; if (gr >= nrows) gr = nrows - 1;
    float4 v = *(const float4*)(ea + (size_t)gr * 32 + c4);
    __bf16* d = sA + r * 40 + c4;
    d[0] = (__bf16)v.x; d[1] = (__bf16)v.y; d[2] = (__bf16)v.z; d[3] = (__bf16)v.w;
  }
  __syncthreads();
  const int wave = tid >> 5, lane = tid & 31, half = lane >> 4, n = lane & 15;
  const bool full = (row0 + 64 <= nrows);
  const int rbase = row0 + wave * 16 + half * 8;
  __bf16* outb = out + (size_t)rbase * 128 + n;
  v16bf afr = frag_a(sA + wave * 16 * 40, 40);
#pragma unroll
  for (int nt = 0; nt < 8; ++nt) {
    const int n0 = nt * 16;
    v8f acc = frag_bias(b, n0);
    acc = WMMA_BF16(afr, frag_b(sWt, 40, 0, n0), acc);
    __bf16* po = outb + n0;
    if (full) {
#pragma unroll
      for (int r = 0; r < 8; ++r) po[(size_t)r * 128] = (__bf16)acc[r];
    } else {
#pragma unroll
      for (int r = 0; r < 8; ++r)
        if (rbase + r < nrows) po[(size_t)r * 128] = (__bf16)acc[r];
    }
  }
}

// vectorized zero-fill: n must be a multiple of 4
__global__ void k_fill4(float4* __restrict__ p, int n4) {
  int i = blockIdx.x * blockDim.x + threadIdx.x;
  if (i < n4) p[i] = make_float4(0.0f, 0.0f, 0.0f, 0.0f);
}

__global__ void k_degree(const int* __restrict__ dst, float* __restrict__ deg, int ne) {
  int e = blockIdx.x * blockDim.x + threadIdx.x;
  if (e < ne) atomicAdd(&deg[dst[e]], 1.0f);
}

// -------- fused: h = gelu((node[src]+eemb)@W1+b1); msg=h@W2+b2; scatter ------
__global__ __launch_bounds__(256) void k_msg(
    const float* __restrict__ node, const __bf16* __restrict__ eemb,
    const int* __restrict__ src, const int* __restrict__ dst,
    const __bf16* __restrict__ w1t, const float* __restrict__ b1,
    const __bf16* __restrict__ w2t, const float* __restrict__ b2,
    float* __restrict__ agg, int nedges) {
  __shared__ __bf16 sW1[128 * 136];
  __shared__ __bf16 sW2[128 * 136];
  __shared__ __bf16 sA[128 * 136];
  __shared__ __bf16 sH[128 * 136];
  const int tid = threadIdx.x;
  const int e0 = blockIdx.x * 128;
  for (int i = tid; i < (128 * 136) / 8; i += 256) {
    ((v8bf*)sW1)[i] = ((const v8bf*)w1t)[i];
    ((v8bf*)sW2)[i] = ((const v8bf*)w2t)[i];
  }
  for (int i = tid; i < 128 * 32; i += 256) {
    int r = i >> 5, c4 = (i & 31) << 2;
    int ge = e0 + r; if (ge >= nedges) ge = nedges - 1;
    int s = src[ge];
    float4 xv = *(const float4*)(node + (size_t)s * 128 + c4);
    v4bf ev = *(const v4bf*)(eemb + (size_t)ge * 128 + c4);
    __bf16* d = sA + r * 136 + c4;
    d[0] = (__bf16)(xv.x + (float)ev[0]);
    d[1] = (__bf16)(xv.y + (float)ev[1]);
    d[2] = (__bf16)(xv.z + (float)ev[2]);
    d[3] = (__bf16)(xv.w + (float)ev[3]);
  }
  __syncthreads();
  const int wave = tid >> 5, lane = tid & 31, half = lane >> 4, n = lane & 15;
  {  // phase 1: H = gelu(A @ W1 + b1)
    const __bf16* at = sA + wave * 16 * 136;
    v16bf afr[4];
#pragma unroll
    for (int kk = 0; kk < 4; ++kk) afr[kk] = frag_a(at + kk * 32, 136);
#pragma unroll
    for (int nt = 0; nt < 8; ++nt) {
      const int n0 = nt * 16;
      v8f acc = frag_bias(b1, n0);
#pragma unroll
      for (int kk = 0; kk < 4; ++kk)
        acc = WMMA_BF16(afr[kk], frag_b(sW1, 136, kk * 32, n0), acc);
#pragma unroll
      for (int r = 0; r < 8; ++r)
        sH[(wave * 16 + half * 8 + r) * 136 + n0 + n] = (__bf16)gelu_exact(acc[r]);
    }
  }
  __syncthreads();
  {  // phase 2: MSG = H @ W2 + b2 ; atomic scatter into agg[dst]
    const __bf16* ht = sH + wave * 16 * 136;
    v16bf hfr[4];
#pragma unroll
    for (int kk = 0; kk < 4; ++kk) hfr[kk] = frag_a(ht + kk * 32, 136);
    const bool full = (e0 + 128 <= nedges);
    int drow[8];
#pragma unroll
    for (int r = 0; r < 8; ++r) {
      int ge = e0 + wave * 16 + half * 8 + r;
      drow[r] = (ge < nedges) ? dst[ge] : -1;
    }
    float* ab = agg + n;
#pragma unroll
    for (int nt = 0; nt < 8; ++nt) {
      const int n0 = nt * 16;
      v8f acc = frag_bias(b2, n0);
#pragma unroll
      for (int kk = 0; kk < 4; ++kk)
        acc = WMMA_BF16(hfr[kk], frag_b(sW2, 136, kk * 32, n0), acc);
      float* ap = ab + n0;
      if (full) {
#pragma unroll
        for (int r = 0; r < 8; ++r)
          atomicAdd(ap + (size_t)drow[r] * 128, acc[r]);
      } else {
#pragma unroll
        for (int r = 0; r < 8; ++r)
          if (drow[r] >= 0) atomicAdd(ap + (size_t)drow[r] * 128, acc[r]);
      }
    }
  }
}

// --- fused: u=gelu((agg/deg)@W1+b1)@W2+b2 ; node=LN(node+u) (per 64 rows) ----
__global__ __launch_bounds__(128) void k_update(
    const float* __restrict__ agg, const float* __restrict__ deg,
    const float* __restrict__ node_in,
    const __bf16* __restrict__ w1t, const float* __restrict__ b1,  // W1^T [256][136]
    const __bf16* __restrict__ w2t, const float* __restrict__ b2,  // W2^T [128][264]
    const float* __restrict__ lng, const float* __restrict__ lnb,
    float* __restrict__ node_out, int nrows) {
  __shared__ __bf16 sW[256 * 136];   // phase1: W1^T ld=136; phase2: W2^T ld=264
  __shared__ __bf16 sA[64 * 136];
  __shared__ __bf16 sU1[64 * 264];
  __shared__ float  sU[64 * 128];
  const int tid = threadIdx.x;
  const int row0 = blockIdx.x * 64;
  for (int i = tid; i < (256 * 136) / 8; i += 128)
    ((v8bf*)sW)[i] = ((const v8bf*)w1t)[i];
  for (int i = tid; i < 64 * 32; i += 128) {    // A = agg / max(deg,1)
    int r = i >> 5, c4 = (i & 31) << 2;
    int gr = row0 + r; if (gr >= nrows) gr = nrows - 1;
    float dg = deg[gr]; if (dg < 1.0f) dg = 1.0f;
    float inv = 1.0f / dg;
    float4 v = *(const float4*)(agg + (size_t)gr * 128 + c4);
    __bf16* d = sA + r * 136 + c4;
    d[0] = (__bf16)(v.x * inv); d[1] = (__bf16)(v.y * inv);
    d[2] = (__bf16)(v.z * inv); d[3] = (__bf16)(v.w * inv);
  }
  __syncthreads();
  const int wave = tid >> 5, lane = tid & 31, half = lane >> 4, n = lane & 15;
  {  // phase 1: U1 = gelu(A @ W1 + b1)  [64 x 256]
    const __bf16* at = sA + wave * 16 * 136;
    v16bf afr[4];
#pragma unroll
    for (int kk = 0; kk < 4; ++kk) afr[kk] = frag_a(at + kk * 32, 136);
#pragma unroll
    for (int nt = 0; nt < 16; ++nt) {
      const int n0 = nt * 16;
      v8f acc = frag_bias(b1, n0);
#pragma unroll
      for (int kk = 0; kk < 4; ++kk)
        acc = WMMA_BF16(afr[kk], frag_b(sW, 136, kk * 32, n0), acc);
#pragma unroll
      for (int r = 0; r < 8; ++r)
        sU1[(wave * 16 + half * 8 + r) * 264 + n0 + n] = (__bf16)gelu_exact(acc[r]);
    }
  }
  __syncthreads();
  for (int i = tid; i < (128 * 264) / 8; i += 128)  // load W2^T over W1^T
    ((v8bf*)sW)[i] = ((const v8bf*)w2t)[i];
  __syncthreads();
  {  // phase 2: U = U1 @ W2 + b2  [64 x 128], K=256
    const __bf16* ut = sU1 + wave * 16 * 264;
    v16bf ufr[8];
#pragma unroll
    for (int kk = 0; kk < 8; ++kk) ufr[kk] = frag_a(ut + kk * 32, 264);
#pragma unroll
    for (int nt = 0; nt < 8; ++nt) {
      const int n0 = nt * 16;
      v8f acc = frag_bias(b2, n0);
#pragma unroll
      for (int kk = 0; kk < 8; ++kk)
        acc = WMMA_BF16(ufr[kk], frag_b(sW, 264, kk * 32, n0), acc);
#pragma unroll
      for (int r = 0; r < 8; ++r)
        sU[(wave * 16 + half * 8 + r) * 128 + n0 + n] = acc[r];
    }
  }
  __syncthreads();
  if (tid < 64) {  // LayerNorm(node_in + U) -> node_out, one row per thread
    int gr = row0 + tid;
    if (gr < nrows) {
      const float4* nr = (const float4*)(node_in + (size_t)gr * 128);
      float4* uv = (float4*)(sU + tid * 128);
      float mean = 0.0f;
#pragma unroll
      for (int c = 0; c < 32; ++c) {   // s = node_in + u, kept in LDS
        float4 a = nr[c], u = uv[c];
        u.x += a.x; u.y += a.y; u.z += a.z; u.w += a.w;
        uv[c] = u;
        mean += u.x + u.y + u.z + u.w;
      }
      mean *= (1.0f / 128.0f);
      float var = 0.0f;
#pragma unroll
      for (int c = 0; c < 32; ++c) {
        float4 u = uv[c];
        float dx = u.x - mean, dy = u.y - mean, dz = u.z - mean, dw = u.w - mean;
        var += dx * dx + dy * dy + dz * dz + dw * dw;
      }
      var *= (1.0f / 128.0f);
      float rstd = rsqrtf(var + 1e-5f);
      const float4* g4 = (const float4*)lng;
      const float4* bb4 = (const float4*)lnb;
      float4* o4 = (float4*)(node_out + (size_t)gr * 128);
#pragma unroll
      for (int c = 0; c < 32; ++c) {
        float4 u = uv[c], g = g4[c], bb = bb4[c], o;
        o.x = (u.x - mean) * rstd * g.x + bb.x;
        o.y = (u.y - mean) * rstd * g.y + bb.y;
        o.z = (u.z - mean) * rstd * g.z + bb.z;
        o.w = (u.w - mean) * rstd * g.w + bb.w;
        o4[c] = o;
      }
    }
  }
}

// ----------------- pooling: global + 7 contiguous segment sums ---------------
// block = 128 threads = 4 row-groups x 32 lanes; lane owns 4 columns (float4)
__global__ __launch_bounds__(128) void k_pool_partial(
    const float* __restrict__ node, float* __restrict__ pool, int nrows) {
  const int tid = threadIdx.x;
  const int lane = tid & 31, q = tid >> 5;
  const int c4 = lane * 4;
  const int base = blockIdx.x * 256;
  const int cs = (nrows + 6) / 7;        // ceil chunk size
  const int seg0 = (base < nrows) ? (base / cs) : 0;
  float4 gs = make_float4(0.f, 0.f, 0.f, 0.f);
  float4 s0 = gs, s1 = gs;
  for (int r = q; r < 256; r += 4) {
    int gr = base + r;
    if (gr >= nrows) break;
    float4 v = *(const float4*)(node + (size_t)gr * 128 + c4);
    gs.x += v.x; gs.y += v.y; gs.z += v.z; gs.w += v.w;
    if (gr / cs == seg0) {
      s0.x += v.x; s0.y += v.y; s0.z += v.z; s0.w += v.w;
    } else {
      s1.x += v.x; s1.y += v.y; s1.z += v.z; s1.w += v.w;
    }
  }
  float* p0 = pool + c4;
  atomicAdd(p0 + 0, gs.x); atomicAdd(p0 + 1, gs.y);
  atomicAdd(p0 + 2, gs.z); atomicAdd(p0 + 3, gs.w);
  float* p1 = pool + (1 + seg0) * 128 + c4;
  atomicAdd(p1 + 0, s0.x); atomicAdd(p1 + 1, s0.y);
  atomicAdd(p1 + 2, s0.z); atomicAdd(p1 + 3, s0.w);
  if (seg0 + 1 <= 6) {
    float* p2 = pool + (2 + seg0) * 128 + c4;
    atomicAdd(p2 + 0, s1.x); atomicAdd(p2 + 1, s1.y);
    atomicAdd(p2 + 2, s1.z); atomicAdd(p2 + 3, s1.w);
  }
}

__global__ void k_pool_final(const float* __restrict__ pool,
                             const float* __restrict__ g, const float* __restrict__ b,
                             float* __restrict__ out, int nrows) {
  int t = threadIdx.x;
  if (t >= 8) return;
  const int cs = (nrows + 6) / 7;
  float cnt;
  if (t == 0) cnt = (float)nrows;
  else {
    int rem = nrows - (t - 1) * cs;
    cnt = (float)(rem < cs ? rem : cs);
  }
  const float inv = 1.0f / cnt;
  float mean = 0.0f;
  for (int c = 0; c < 128; ++c) mean += pool[t * 128 + c] * inv;
  mean *= (1.0f / 128.0f);
  float var = 0.0f;
  for (int c = 0; c < 128; ++c) {
    float d = pool[t * 128 + c] * inv - mean;
    var += d * d;
  }
  var *= (1.0f / 128.0f);
  float rstd = rsqrtf(var + 1e-5f);
  for (int c = 0; c < 128; ++c)
    out[t * 128 + c] = (pool[t * 128 + c] * inv - mean) * rstd * g[c] + b[c];
}

// -----------------------------------------------------------------------------
extern "C" void kernel_launch(void* const* d_in, const int* in_sizes, int n_in,
                              void* d_out, int out_size, void* d_ws, size_t ws_size,
                              hipStream_t stream) {
  const float* x      = (const float*)d_in[0];
  const int*   eidx   = (const int*)d_in[1];     // [2,E] int32: row0=src, row1=dst
  const float* eattr  = (const float*)d_in[2];
  const float* node_w = (const float*)d_in[3];
  const float* node_b = (const float*)d_in[4];
  const float* edge_w = (const float*)d_in[5];
  const float* edge_b = (const float*)d_in[6];
  const float* msg_w1 = (const float*)d_in[7];
  const float* msg_b1 = (const float*)d_in[8];
  const float* msg_w2 = (const float*)d_in[9];
  const float* msg_b2 = (const float*)d_in[10];
  const float* upd_w1 = (const float*)d_in[11];
  const float* upd_b1 = (const float*)d_in[12];
  const float* upd_w2 = (const float*)d_in[13];
  const float* upd_b2 = (const float*)d_in[14];
  const float* ln_g   = (const float*)d_in[15];
  const float* ln_b   = (const float*)d_in[16];
  const float* out_g  = (const float*)d_in[17];
  const float* out_b  = (const float*)d_in[18];
  (void)in_sizes; (void)n_in; (void)out_size; (void)ws_size;

  const int* src = eidx;
  const int* dst = eidx + N_EDGES;

  // ---------------- workspace layout (256B-aligned chunks) -------------------
  char* ws = (char*)d_ws;
  float* node = (float*)ws;   ws += ((size_t)N_NODES * DIM * 4 + 255) & ~(size_t)255;
  float* agg  = (float*)ws;   ws += ((size_t)N_NODES * DIM * 4 + 255) & ~(size_t)255;
  float* deg  = (float*)ws;   ws += ((size_t)N_NODES * 4 + 255) & ~(size_t)255;
  __bf16* eemb = (__bf16*)ws; ws += ((size_t)N_EDGES * DIM * 2 + 255) & ~(size_t)255;
  float* pool = (float*)ws;   ws += (8 * DIM * 4 + 255) & ~(size_t)255;
  // padded transposed bf16 weight images (match LDS layouts exactly)
  __bf16* nodeWT = (__bf16*)ws; ws += ((size_t)128 * 136 * 2 + 255) & ~(size_t)255;
  __bf16* edgeWT = (__bf16*)ws; ws += ((size_t)128 * 40 * 2 + 255) & ~(size_t)255;
  __bf16* msgW1T[NLAYERS], *msgW2T[NLAYERS], *updW1T[NLAYERS], *updW2T[NLAYERS];
  for (int l = 0; l < NLAYERS; ++l) {
    msgW1T[l] = (__bf16*)ws; ws += ((size_t)128 * 136 * 2 + 255) & ~(size_t)255;
    msgW2T[l] = (__bf16*)ws; ws += ((size_t)128 * 136 * 2 + 255) & ~(size_t)255;
    updW1T[l] = (__bf16*)ws; ws += ((size_t)256 * 136 * 2 + 255) & ~(size_t)255;
    updW2T[l] = (__bf16*)ws; ws += ((size_t)128 * 264 * 2 + 255) & ~(size_t)255;
  }

  // ---------------- weight prep (tiny, one-time) -----------------------------
  k_prep_wt<<<(128 * 128 + 255) / 256, 256, 0, stream>>>(node_w, nodeWT, 128, 7, 136);
  k_prep_wt<<<(32 * 128 + 255) / 256, 256, 0, stream>>>(edge_w, edgeWT, 32, 7, 40);
  for (int l = 0; l < NLAYERS; ++l) {
    k_prep_wt<<<(128 * 128 + 255) / 256, 256, 0, stream>>>(
        msg_w1 + (size_t)l * DIM * DIM, msgW1T[l], 128, 7, 136);
    k_prep_wt<<<(128 * 128 + 255) / 256, 256, 0, stream>>>(
        msg_w2 + (size_t)l * DIM * DIM, msgW2T[l], 128, 7, 136);
    k_prep_wt<<<(128 * 256 + 255) / 256, 256, 0, stream>>>(
        upd_w1 + (size_t)l * DIM * 2 * DIM, updW1T[l], 128, 8, 136);
    k_prep_wt<<<(256 * 128 + 255) / 256, 256, 0, stream>>>(
        upd_w2 + (size_t)l * 2 * DIM * DIM, updW2T[l], 256, 7, 264);
  }

  // ---------------- pipeline -------------------------------------------------
  k_node_embed<<<(N_NODES + 63) / 64, 128, 0, stream>>>(x, nodeWT, node_b, node, N_NODES);
  k_edge_embed<<<(N_EDGES + 63) / 64, 128, 0, stream>>>(eattr, edgeWT, edge_b, eemb, N_EDGES);
  k_fill4<<<(N_NODES / 4 + 255) / 256, 256, 0, stream>>>((float4*)deg, N_NODES / 4);
  k_degree<<<(N_EDGES + 255) / 256, 256, 0, stream>>>(dst, deg, N_EDGES);

  for (int l = 0; l < NLAYERS; ++l) {
    k_fill4<<<(N_NODES * (DIM / 4) + 255) / 256, 256, 0, stream>>>(
        (float4*)agg, N_NODES * (DIM / 4));
    k_msg<<<(N_EDGES + 127) / 128, 256, 0, stream>>>(
        node, eemb, src, dst,
        msgW1T[l], msg_b1 + l * DIM,
        msgW2T[l], msg_b2 + l * DIM, agg, N_EDGES);
    k_update<<<(N_NODES + 63) / 64, 128, 0, stream>>>(
        agg, deg, node,
        updW1T[l], upd_b1 + l * 2 * DIM,
        updW2T[l], upd_b2 + l * DIM,
        ln_g + l * DIM, ln_b + l * DIM, node, N_NODES);
  }

  k_fill4<<<(8 * DIM / 4 + 255) / 256, 256, 0, stream>>>((float4*)pool, 8 * DIM / 4);
  k_pool_partial<<<(N_NODES + 255) / 256, 128, 0, stream>>>(node, pool, N_NODES);
  k_pool_final<<<1, 32, 0, stream>>>(pool, out_g, out_b, (float*)d_out, N_NODES);
}
